// MyModel_57415122813720
// MI455X (gfx1250) — matrix operand order repaired and checked by reading, not compiled, not run on hardware
//
#include <hip/hip_runtime.h>
#include <hip/hip_bf16.h>
#include <math.h>

#if __has_include(<hip/amd_detail/amd_gfx1250_TDM.h>)
#define TDM_SIX_ARGS 1
#endif

typedef __attribute__((ext_vector_type(16))) __bf16 bf16x16;
typedef __attribute__((ext_vector_type(8)))  __bf16 bf16x8;
typedef __attribute__((ext_vector_type(8)))  float  f32x8;
typedef __attribute__((ext_vector_type(4)))  unsigned int uint32x4;
typedef __attribute__((ext_vector_type(8)))  int int32x8;
typedef __attribute__((ext_vector_type(4)))  int int32x4;

constexpr int B_    = 16384;
constexpr int H_    = 32;
constexpr int KVH_  = 8;
constexpr int D_    = 128;
constexpr int HID_  = 4096;      // H*D
constexpr int KVW_  = 1024;      // KVH*D
constexpr int RBIT_ = 128;
constexpr float INV_SQRT_D = 0.08838834764831845f;

constexpr int MBLK = 128;        // batch rows per block
constexpr int KCH  = 32;         // K chunk
constexpr int NT   = 256;        // threads per block (8 waves)
constexpr int KT   = HID_ / KCH; // 128 chunks

// workspace byte offsets
constexpr size_t WS_KST   = 0;           // k_states fp32 [B][KVH*D]   = 64 MB
constexpr size_t WS_WQ_HI = 67108864;    // q_w^T hi bf16 [4096][4096] = 32 MB
constexpr size_t WS_WQ_LO = 100663296;   // q_w^T lo bf16              = 32 MB
constexpr size_t WS_WK_HI = 134217728;   // k_w^T hi bf16 [1024][4096] =  8 MB
constexpr size_t WS_WK_LO = 142606336;   // k_w^T lo bf16              =  8 MB

__device__ __forceinline__ void bsplit(float x, __bf16* ph, __bf16* pl) {
  __bf16 h = (__bf16)x;
  *ph = h;
  *pl = (__bf16)(x - (float)h);
}

__device__ __forceinline__ bf16x16 load_frag(const __bf16* p0, const __bf16* p1) {
  bf16x8 r0 = *(const bf16x8*)p0;
  bf16x8 r1 = *(const bf16x8*)p1;
  return __builtin_shufflevector(r0, r1, 0,1,2,3,4,5,6,7,8,9,10,11,12,13,14,15);
}

__device__ __forceinline__ f32x8 wmma_bf16(bf16x16 a, bf16x16 b, f32x8 c) {
  return __builtin_amdgcn_wmma_f32_16x16x32_bf16(false, a, false, b, (short)0, c,
                                                 false, false);
}

// ---- Tensor Data Mover: 2D tile (tile_rows x 32 bf16, row stride 4096 elems)
// from global -> LDS. D# bit layout per CDNA5 ISA 8.3/8.4:
//   g0: [1:0]=count=1, [63:32]=lds_addr, [120:64]=global_addr, [127:126]=type=2
//   g1: data_size=2B, tensor_dim0/1 = 1<<20 (no clipping), tile_dim0=32,
//       tile_dim1=tile_rows, tensor_dim0_stride=4096
__device__ __forceinline__ void tdm_load_tile(unsigned int lds_off,
                                              const void* gptr,
                                              int tile_rows) {
  unsigned long long ga = (unsigned long long)(uintptr_t)gptr;
  uint32x4 g0;
  g0.x = 1u;
  g0.y = (unsigned int)__builtin_amdgcn_readfirstlane((int)lds_off);
  g0.z = (unsigned int)__builtin_amdgcn_readfirstlane((int)(unsigned int)ga);
  g0.w = (unsigned int)__builtin_amdgcn_readfirstlane(
           (int)((((unsigned int)(ga >> 32)) & 0x01FFFFFFu) | 0x80000000u));
  int32x8 g1;
  g1[0] = 0x00010000;   // workgroup_mask=0, data_size=1 (2 bytes)
  g1[1] = 0x00000000;   // tensor_dim0 low16 << 16 (1<<20 -> 0)
  g1[2] = 0x00000010;   // tensor_dim0 hi16 | tensor_dim1 low16 << 16
  g1[3] = 0x00200010;   // tensor_dim1 hi16 | tile_dim0(32) << 16
  g1[4] = tile_rows;    // tile_dim1 | tile_dim2(0) << 16
  g1[5] = 4096;         // tensor_dim0_stride low32 (elements)
  g1[6] = 0;
  g1[7] = 0;
  int32x4 z4 = {0, 0, 0, 0};
#ifdef TDM_SIX_ARGS
  int32x8 z8 = {0, 0, 0, 0, 0, 0, 0, 0};
  __builtin_amdgcn_tensor_load_to_lds(g0, g1, z4, z4, z8, 0);
#else
  __builtin_amdgcn_tensor_load_to_lds(g0, g1, z4, z4, 0);
#endif
}

// ---- A-side chunk fill: fp32 global -> hi/lo bf16 LDS [m][k]
__device__ __forceinline__ void fillA(const float* __restrict__ X, int row0,
                                      int k0, __bf16* Ah, __bf16* Al, int tid) {
  #pragma unroll
  for (int i = 0; i < 4; ++i) {
    int f   = i * NT + tid;
    int row = f >> 3;
    int kq  = (f & 7) << 2;
    const float4 v = *(const float4*)(X + (size_t)(row0 + row) * HID_ + k0 + kq);
    __bf16* ah = Ah + row * KCH + kq;
    __bf16* al = Al + row * KCH + kq;
    bsplit(v.x, ah + 0, al + 0);
    bsplit(v.y, ah + 1, al + 1);
    bsplit(v.z, ah + 2, al + 2);
    bsplit(v.w, ah + 3, al + 3);
  }
}

// Block GEMM: C[128x128] = X[row0:+128, :4096] * W'[n0:+128, :4096]^T
// W' is pre-split/pre-transposed bf16 hi/lo [n][k]. Double-buffered LDS:
//   A buffers at p*16384 (hi 0..8K, lo 8K..16K), B at 32768 + p*16384.
// B tiles fetched by TDM (2 ops/wave/chunk, 16 rows each), A filled by VALU.
__device__ __forceinline__ void gemm_block(
    const float* __restrict__ X,
    const __bf16* __restrict__ WhiT, const __bf16* __restrict__ WloT,
    int n0, int row0, char* smbase, unsigned int ldsbase, f32x8 acc[8])
{
  const int tid  = threadIdx.x;
  const int wv   = tid >> 5;
  const int lane = tid & 31;
  const int hf   = lane >> 4;
  const int ln   = lane & 15;

  // prologue: stage chunk 0
  fillA(X, row0, 0, (__bf16*)smbase, (__bf16*)(smbase + 8192), tid);
  {
    unsigned int ldsB = ldsbase + 32768u + (unsigned)(wv * 16 * KCH * 2);
    const __bf16* ghi = WhiT + (size_t)(n0 + wv * 16) * HID_;
    const __bf16* glo = WloT + (size_t)(n0 + wv * 16) * HID_;
    tdm_load_tile(ldsB, ghi, 16);
    tdm_load_tile(ldsB + 8192u, glo, 16);
  }

  for (int kc = 0; kc < KT; ++kc) {
    const int p = kc & 1;
    __syncthreads();   // all waves done reading buffers 1-p
    if (kc + 1 < KT) {
      const int k0n = (kc + 1) * KCH;
      fillA(X, row0, k0n, (__bf16*)(smbase + (1 - p) * 16384),
            (__bf16*)(smbase + (1 - p) * 16384 + 8192), tid);
      unsigned int ldsB = ldsbase + 32768u + (unsigned)((1 - p) * 16384)
                        + (unsigned)(wv * 16 * KCH * 2);
      const __bf16* ghi = WhiT + (size_t)(n0 + wv * 16) * HID_ + k0n;
      const __bf16* glo = WloT + (size_t)(n0 + wv * 16) * HID_ + k0n;
      tdm_load_tile(ldsB, ghi, 16);
      tdm_load_tile(ldsB + 8192u, glo, 16);
      __builtin_amdgcn_s_wait_tensorcnt(2);  // chunk kc's pair complete
    } else {
      __builtin_amdgcn_s_wait_tensorcnt(0);
    }
    __syncthreads();   // buffers p ready for everyone

    const __bf16* Ah = (const __bf16*)(smbase + p * 16384);
    const __bf16* Al = Ah + 4096;
    const __bf16* Bh = (const __bf16*)(smbase + 32768 + p * 16384);
    const __bf16* Bl = Bh + 4096;

    const __bf16* arh = Ah + (wv * 16 + ln) * KCH;
    const __bf16* arl = Al + (wv * 16 + ln) * KCH;
    bf16x16 a_h = load_frag(arh + hf * 8, arh + 16 + hf * 8);
    bf16x16 a_l = load_frag(arl + hf * 8, arl + 16 + hf * 8);
    #pragma unroll
    for (int t = 0; t < 8; ++t) {
      const __bf16* brh = Bh + (t * 16 + ln) * KCH + hf * 8;
      const __bf16* brl = Bl + (t * 16 + ln) * KCH + hf * 8;
      bf16x16 b_h = load_frag(brh, brh + 16);
      bf16x16 b_l = load_frag(brl, brl + 16);
      acc[t] = wmma_bf16(a_h, b_h, acc[t]);
      acc[t] = wmma_bf16(a_h, b_l, acc[t]);
      acc[t] = wmma_bf16(a_l, b_h, acc[t]);
    }
  }
}

// ---- one-shot weight preprocessing: W[k][n] fp32 -> W'[n][k] hi/lo bf16 ----
__global__ void prep_w_kernel(const float* __restrict__ W, int N,
                              __bf16* __restrict__ Whi, __bf16* __restrict__ Wlo)
{
  __shared__ float tile[64][65];
  const int tk  = blockIdx.x * 64;
  const int tn  = blockIdx.y * 64;
  const int tid = threadIdx.x;
  #pragma unroll
  for (int i = 0; i < 4; ++i) {
    int f = i * 256 + tid;
    int r = f >> 4;
    int c = (f & 15) << 2;
    const float4 v = *(const float4*)(W + (size_t)(tk + r) * N + tn + c);
    tile[r][c + 0] = v.x; tile[r][c + 1] = v.y;
    tile[r][c + 2] = v.z; tile[r][c + 3] = v.w;
  }
  __syncthreads();
  const int nl = tid >> 2;
  const int ks = (tid & 3) << 4;
  const size_t obase = (size_t)(tn + nl) * HID_ + tk + ks;
  #pragma unroll
  for (int g = 0; g < 2; ++g) {
    bf16x8 vh, vl;
    #pragma unroll
    for (int j = 0; j < 8; ++j) {
      float x = tile[ks + g * 8 + j][nl];
      __bf16 h = (__bf16)x;
      vh[j] = h;
      vl[j] = (__bf16)(x - (float)h);
    }
    *(bf16x8*)(Whi + obase + g * 8) = vh;
    *(bf16x8*)(Wlo + obase + g * 8) = vl;
  }
}

// ---------------- K projection: k_states -> workspace ----------------
__global__ void k_proj_kernel(const float* __restrict__ k_hidden,
                              const __bf16* __restrict__ wk_hi,
                              const __bf16* __restrict__ wk_lo,
                              const float* __restrict__ k_b,
                              float* __restrict__ kws)
{
  extern __shared__ char smem[];
  const unsigned int ldsbase = (unsigned int)(uintptr_t)&smem[0];

  const int row0    = blockIdx.x * MBLK;
  const int kvh     = blockIdx.y;
  const int colbase = kvh * D_;

  f32x8 zero = {0.f, 0.f, 0.f, 0.f, 0.f, 0.f, 0.f, 0.f};
  f32x8 acc[8];
  #pragma unroll
  for (int t = 0; t < 8; ++t) acc[t] = zero;

  gemm_block(k_hidden, wk_hi, wk_lo, colbase, row0, smem, ldsbase, acc);

  const int tid = threadIdx.x;
  const int wv = tid >> 5, lane = tid & 31, hf = lane >> 4, ln = lane & 15;
  #pragma unroll
  for (int t = 0; t < 8; ++t) {
    const int n = colbase + t * 16 + ln;
    const float bias = k_b[n];
    #pragma unroll
    for (int r = 0; r < 8; ++r) {
      const int m = wv * 16 + hf * 8 + r;
      kws[(size_t)(row0 + m) * KVW_ + n] = acc[t][r] + bias;
    }
  }
}

// ------- Q projection + fused attn/hash/hamming epilogue -------
__global__ void q_fused_kernel(const float* __restrict__ q_hidden,
                               const __bf16* __restrict__ wq_hi,
                               const __bf16* __restrict__ wq_lo,
                               const float* __restrict__ q_b,
                               const float* __restrict__ hash_w,
                               const float* __restrict__ kws,
                               float* __restrict__ out)
{
  extern __shared__ char smem[];
  const unsigned int ldsbase = (unsigned int)(uintptr_t)&smem[0];

  __bf16* Wh2  = (__bf16*)(smem);              // hash_w hi [r][d], reuses gemm region
  __bf16* Wl2  = (__bf16*)(smem + 32768);      // hash_w lo
  __bf16* q_sh = (__bf16*)(smem + 65536);      // q_states hi [128][128]
  __bf16* q_sl = (__bf16*)(smem + 98304);      // q_states lo
  float*  normw = (float*)(smem + 131072);     // per-row qn*kn*inv_sqrt_d

  const int row0    = blockIdx.x * MBLK;
  const int h       = blockIdx.y;
  const int kvh     = h >> 2;                  // G = 4
  const int colbase = h * D_;

  const int tid = threadIdx.x;
  const int wv = tid >> 5, lane = tid & 31, hf = lane >> 4, ln = lane & 15;

  f32x8 zero = {0.f, 0.f, 0.f, 0.f, 0.f, 0.f, 0.f, 0.f};
  f32x8 acc[8];
  #pragma unroll
  for (int t = 0; t < 8; ++t) acc[t] = zero;

  gemm_block(q_hidden, wq_hi, wq_lo, colbase, row0, smem, ldsbase, acc);

  __syncthreads();  // gemm LDS region now reusable

  // bias add, dump q_states tile to LDS as hi/lo bf16
  #pragma unroll
  for (int t = 0; t < 8; ++t) {
    const float bias = q_b[colbase + t * 16 + ln];
    #pragma unroll
    for (int r = 0; r < 8; ++r) {
      const int m = wv * 16 + hf * 8 + r;
      const int n = t * 16 + ln;
      bsplit(acc[t][r] + bias, &q_sh[m * D_ + n], &q_sl[m * D_ + n]);
    }
  }

  // stage hash_w [d][r] -> transposed bf16 hi/lo [r][d]
  #pragma unroll
  for (int i = 0; i < 16; ++i) {
    int f  = i * NT + tid;
    int d  = f >> 5;
    int rq = (f & 31) << 2;
    const float4 v = *(const float4*)(hash_w + (size_t)d * RBIT_ + rq);
    bsplit(v.x, &Wh2[(rq + 0) * D_ + d], &Wl2[(rq + 0) * D_ + d]);
    bsplit(v.y, &Wh2[(rq + 1) * D_ + d], &Wl2[(rq + 1) * D_ + d]);
    bsplit(v.z, &Wh2[(rq + 2) * D_ + d], &Wl2[(rq + 2) * D_ + d]);
    bsplit(v.w, &Wh2[(rq + 3) * D_ + d], &Wl2[(rq + 3) * D_ + d]);
  }
  __syncthreads();

  // scaled dot + norms (one thread per row, waves 0..3)
  if (tid < MBLK) {
    const int m = tid;
    const __bf16* qh = q_sh + m * D_;
    const __bf16* ql = q_sl + m * D_;
    const float* kr = kws + (size_t)(row0 + m) * KVW_ + kvh * D_;
    float dot = 0.f, qn = 0.f, kn = 0.f;
    for (int d = 0; d < D_; ++d) {
      float qv = (float)qh[d] + (float)ql[d];
      float kv = kr[d];
      dot += qv * kv;
      qn  += qv * qv;
      kn  += kv * kv;
    }
    out[(size_t)(row0 + m) * H_ + h] = dot * INV_SQRT_D;
    normw[m] = sqrtf(qn) * sqrtf(kn) * INV_SQRT_D;
  }

  // hash GEMMs (q and k codes) via WMMA, K = 128 in 4 chunks
  f32x8 aq[8], ak[8];
  #pragma unroll
  for (int t = 0; t < 8; ++t) { aq[t] = zero; ak[t] = zero; }

  const float* krow = kws + (size_t)(row0 + wv * 16 + ln) * KVW_ + kvh * D_;
  #pragma unroll
  for (int kc = 0; kc < 4; ++kc) {
    const int k0 = kc * 32;
    const __bf16* qrh = q_sh + (wv * 16 + ln) * D_ + k0;
    const __bf16* qrl = q_sl + (wv * 16 + ln) * D_ + k0;
    bf16x16 qa_h = load_frag(qrh + hf * 8, qrh + 16 + hf * 8);
    bf16x16 qa_l = load_frag(qrl + hf * 8, qrl + 16 + hf * 8);

    bf16x16 ka_h = {}, ka_l = {};
    #pragma unroll
    for (int j = 0; j < 8; ++j) {
      float x0 = krow[k0 + hf * 8 + j];
      float x1 = krow[k0 + 16 + hf * 8 + j];
      __bf16 h0 = (__bf16)x0;
      ka_h[j]     = h0;  ka_l[j]     = (__bf16)(x0 - (float)h0);
      __bf16 h1 = (__bf16)x1;
      ka_h[8 + j] = h1;  ka_l[8 + j] = (__bf16)(x1 - (float)h1);
    }

    #pragma unroll
    for (int t = 0; t < 8; ++t) {
      const __bf16* brh = Wh2 + (t * 16 + ln) * D_ + k0 + hf * 8;
      const __bf16* brl = Wl2 + (t * 16 + ln) * D_ + k0 + hf * 8;
      bf16x16 b_h = load_frag(brh, brh + 16);
      bf16x16 b_l = load_frag(brl, brl + 16);
      aq[t] = wmma_bf16(qa_h, b_h, aq[t]);
      aq[t] = wmma_bf16(qa_h, b_l, aq[t]);
      aq[t] = wmma_bf16(qa_l, b_h, aq[t]);
      ak[t] = wmma_bf16(ka_h, b_h, ak[t]);
      ak[t] = wmma_bf16(ka_h, b_l, ak[t]);
      ak[t] = wmma_bf16(ka_l, b_h, ak[t]);
    }
  }

  // hamming distance: |(tanh(q)+1)/2 - (tanh(k)+1)/2| summed over RBIT
  float hp[8];
  #pragma unroll
  for (int r = 0; r < 8; ++r) hp[r] = 0.f;
  #pragma unroll
  for (int t = 0; t < 8; ++t) {
    #pragma unroll
    for (int r = 0; r < 8; ++r)
      hp[r] += 0.5f * fabsf(tanhf(aq[t][r]) - tanhf(ak[t][r]));
  }
  // reduce across the 16 lanes sharing the same rows (wave32, xor < 16)
  #pragma unroll
  for (int msk = 1; msk < 16; msk <<= 1) {
    #pragma unroll
    for (int r = 0; r < 8; ++r) hp[r] += __shfl_xor(hp[r], msk, 32);
  }

  __syncthreads();  // normw visible to all waves

  if (ln == 0) {
    #pragma unroll
    for (int r = 0; r < 8; ++r) {
      const int m = wv * 16 + hf * 8 + r;
      out[(size_t)B_ * H_ + (size_t)(row0 + m) * H_ + h] =
          (1.0f - hp[r] * (1.0f / 64.0f)) * normw[m];
    }
  }
}

extern "C" void kernel_launch(void* const* d_in, const int* in_sizes, int n_in,
                              void* d_out, int out_size, void* d_ws, size_t ws_size,
                              hipStream_t stream)
{
  const float* q_hidden = (const float*)d_in[0];
  const float* k_hidden = (const float*)d_in[1];
  const float* q_w      = (const float*)d_in[2];
  const float* q_b      = (const float*)d_in[3];
  const float* k_w      = (const float*)d_in[4];
  const float* k_b      = (const float*)d_in[5];
  const float* hash_w   = (const float*)d_in[6];
  float* out = (float*)d_out;

  char* ws = (char*)d_ws;
  float*  kws   = (float*)(ws + WS_KST);
  __bf16* wq_hi = (__bf16*)(ws + WS_WQ_HI);
  __bf16* wq_lo = (__bf16*)(ws + WS_WQ_LO);
  __bf16* wk_hi = (__bf16*)(ws + WS_WK_HI);
  __bf16* wk_lo = (__bf16*)(ws + WS_WK_LO);

  (void)in_sizes; (void)n_in; (void)out_size; (void)ws_size;

  dim3 blk(NT);
  // one-shot weight preprocessing (L2-resident afterwards)
  hipLaunchKernelGGL(prep_w_kernel, dim3(HID_ / 64, HID_ / 64), blk, 0, stream,
                     q_w, HID_, wq_hi, wq_lo);
  hipLaunchKernelGGL(prep_w_kernel, dim3(HID_ / 64, KVW_ / 64), blk, 0, stream,
                     k_w, KVW_, wk_hi, wk_lo);
  // projections
  hipLaunchKernelGGL(k_proj_kernel, dim3(B_ / MBLK, KVH_), blk, 65536, stream,
                     k_hidden, wk_hi, wk_lo, k_b, kws);
  hipLaunchKernelGGL(q_fused_kernel, dim3(B_ / MBLK, H_), blk, 131584, stream,
                     q_hidden, wq_hi, wq_lo, q_b, hash_w, kws, out);
}
